// MSAColumnGlobalAttention_22814866276377
// MI455X (gfx1250) — matrix operand order repaired and checked by reading, not compiled, not run on hardware
//
#include <hip/hip_runtime.h>

#define DEV __device__ __forceinline__

typedef __attribute__((ext_vector_type(16))) __bf16 v16bf;
typedef __attribute__((ext_vector_type(2)))  __bf16 bf16x2;
typedef __attribute__((ext_vector_type(8)))  float  v8f;

constexpr int S  = 2048;   // N_seq
constexpr int R  = 384;    // N_res
constexpr int CM = 64;     // c_m
constexpr int TILES = (S * R) / 16;        // 49152 row-tiles of 16
constexpr int WPB = 8;                     // waves per block (256 threads)
constexpr int TPW = 8;                     // tiles per wave
constexpr int GEMM_BLOCKS = TILES / (WPB * TPW);  // 768

DEV v8f wmma_bf16(v16bf a, v16bf b, v8f c) {
  // D = A(16x32 bf16) x B(32x16 bf16) + C(16x16 f32)
  return __builtin_amdgcn_wmma_f32_16x16x32_bf16(false, a, false, b, (short)0, c,
                                                 false, false);
}

// LayerNorm over c_m=64 with 2 lanes per row: lane half h owns columns 32h..32h+31.
DEV void ln_row_half(const float* __restrict__ mrow, const float* gs, const float* bs,
                     int h, float x[32]) {
  float4 f[8];
#pragma unroll
  for (int j = 0; j < 8; ++j) f[j] = ((const float4*)mrow)[j];
  float s = 0.f, s2 = 0.f;
#pragma unroll
  for (int j = 0; j < 8; ++j) {
    s  += f[j].x + f[j].y + f[j].z + f[j].w;
    s2 += f[j].x * f[j].x + f[j].y * f[j].y + f[j].z * f[j].z + f[j].w * f[j].w;
  }
  s  += __shfl_xor(s, 16);
  s2 += __shfl_xor(s2, 16);
  float mu   = s * (1.f / 64.f);
  float var  = s2 * (1.f / 64.f) - mu * mu;
  float rstd = rsqrtf(var + 1e-5f);
#pragma unroll
  for (int j = 0; j < 8; ++j) {
    int c = 32 * h + 4 * j;
    x[4*j+0] = (f[j].x - mu) * rstd * gs[c+0] + bs[c+0];
    x[4*j+1] = (f[j].y - mu) * rstd * gs[c+1] + bs[c+1];
    x[4*j+2] = (f[j].z - mu) * rstd * gs[c+2] + bs[c+2];
    x[4*j+3] = (f[j].w - mu) * rstd * gs[c+3] + bs[c+3];
  }
}

// Stage this lane's 32 columns of its row as bf16 pairs into LDS (row stride 66).
DEV void stage_x(__bf16* xrow, int h, const float x[32]) {
#pragma unroll
  for (int j = 0; j < 16; ++j) {
    bf16x2 p;
    p[0] = (__bf16)x[2*j];
    p[1] = (__bf16)x[2*j+1];
    *(bf16x2*)(xrow + 32 * h + 2 * j) = p;
  }
}

// A-fragment (16x32 bf16): row = lane&15; lanes<16: K {0..15},
// lanes>=16: K {16..31}; elements 2j,2j+1 are consecutive K (one VGPR).
DEV v16bf load_a_frag(const __bf16* xrow, int h, int kc) {
  v16bf a;
  const int b0 = 32 * kc + 16 * h;
#pragma unroll
  for (int j = 0; j < 4; ++j) {
    bf16x2 p0 = *(const bf16x2*)(xrow + b0 + 2 * j);
    a[2*j]   = p0[0];
    a[2*j+1] = p0[1];
    bf16x2 p1 = *(const bf16x2*)(xrow + b0 + 8 + 2 * j);
    a[8+2*j]   = p1[0];
    a[8+2*j+1] = p1[1];
  }
  return a;
}

// ---------------------------------------------------------------------------
// Pass 1: LayerNorm + k = x@Wk, v = x@Wv (single WMMA with B = [Wk|Wv] 64x16),
// plus deterministic per-16-seq-tile column sums of x (for q_mean).
// Tile gt -> residue r = gt>>7, seq-tile st = gt&127 (16 consecutive s, fixed r).
// ---------------------------------------------------------------------------
__global__ __launch_bounds__(256)
void msa_pass1(const float* __restrict__ m, const float* __restrict__ gamma,
               const float* __restrict__ beta, const float* __restrict__ Wk,
               const float* __restrict__ Wv, float* __restrict__ kbuf,
               float* __restrict__ vbuf, float* __restrict__ partials) {
  __shared__ float  gs[64], bs[64];
  __shared__ __bf16 xs[WPB * 16 * 66];
  const int t = threadIdx.x;
  if (t < 64) { gs[t] = gamma[t]; bs[t] = beta[t]; }
  __syncthreads();

  const int lane = t & 31, wv = t >> 5;
  const int h = lane >> 4, rl = lane & 15, n = rl;

  // B fragments for fused [Wk | Wv]: cols 0..7 -> k, 8..15 -> v.
  v16bf bkv[2];
#pragma unroll
  for (int kc = 0; kc < 2; ++kc) {
#pragma unroll
    for (int e = 0; e < 16; ++e) {
      int K = 32 * kc + 16 * h + e;
      float w = (n < 8) ? Wk[K * 8 + n] : Wv[K * 8 + (n - 8)];
      bkv[kc][e] = (__bf16)w;
    }
  }

  __bf16* xrow = xs + (wv * 16 + rl) * 66;

  for (int t8 = 0; t8 < TPW; ++t8) {
    int gt = (blockIdx.x * WPB + wv) * TPW + t8;
    int r = gt >> 7, st = gt & 127;
    int s0 = st * 16;

    const float* mrow = m + ((size_t)(s0 + rl) * R + r) * CM + 32 * h;
    float x[32];
    ln_row_half(mrow, gs, bs, h, x);

    // Column sums over the 16 seq rows of this tile (butterfly within 16-lane half).
    float p[32];
#pragma unroll
    for (int j = 0; j < 32; ++j) p[j] = x[j];
#pragma unroll
    for (int d = 1; d <= 8; d <<= 1) {
#pragma unroll
      for (int j = 0; j < 32; ++j) p[j] += __shfl_xor(p[j], d);
    }
    if (rl == 0) {
      float* pd = partials + ((size_t)r * 128 + st) * 64 + 32 * h;
#pragma unroll
      for (int j = 0; j < 32; ++j) pd[j] = p[j];
    }

    stage_x(xrow, h, x);
    v16bf a0 = load_a_frag(xrow, h, 0);
    v16bf a1 = load_a_frag(xrow, h, 1);

    v8f acc = {0.f, 0.f, 0.f, 0.f, 0.f, 0.f, 0.f, 0.f};
    acc = wmma_bf16(a0, bkv[0], acc);
    acc = wmma_bf16(a1, bkv[1], acc);

    // D layout: row = i + 8h, col = n.  Store k/v in [r][s][c] layout.
    float* base = (n < 8)
        ? (kbuf + ((size_t)r * S + s0 + 8 * h) * 8 + n)
        : (vbuf + ((size_t)r * S + s0 + 8 * h) * 8 + (n - 8));
#pragma unroll
    for (int i = 0; i < 8; ++i) base[(size_t)i * 8] = acc[i];
  }
}

// ---------------------------------------------------------------------------
// Attention: per residue r. q = (mean_s x) @ Wq / sqrt(8); softmax over S;
// o = a @ v.  One 32-lane group per head; logits live in registers (64/lane).
// ---------------------------------------------------------------------------
__global__ __launch_bounds__(256)
void msa_attn(const float* __restrict__ partials, const float* __restrict__ Wq,
              const float* __restrict__ kbuf, const float* __restrict__ vbuf,
              float* __restrict__ obuf) {
  const int r = blockIdx.x, t = threadIdx.x;
  __shared__ float qsum_s[64];
  __shared__ float qv[64];

  if (t < 64) {
    float acc = 0.f;
    const float* p = partials + (size_t)r * 128 * 64 + t;
    for (int st = 0; st < 128; ++st) acc += p[(size_t)st * 64];
    qsum_s[t] = acc;
  }
  __syncthreads();
  if (t < 64) {
    float acc = 0.f;
#pragma unroll
    for (int c = 0; c < 64; ++c) acc += qsum_s[c] * Wq[c * 64 + t];
    qv[t] = acc * (1.f / (float)S) * 0.3535533905932738f;  // 1/sqrt(8)
  }
  __syncthreads();

  const int h = t >> 5, lane = t & 31;
  float qh[8];
#pragma unroll
  for (int c = 0; c < 8; ++c) qh[c] = qv[h * 8 + c];

  const float* kb = kbuf + (size_t)r * S * 8;
  const float* vb = vbuf + (size_t)r * S * 8;

  float ls[64];
  float mx = -3.0e38f;
#pragma unroll
  for (int j = 0; j < 64; ++j) {
    int s = lane + 32 * j;
    const float4* kp = (const float4*)(kb + (size_t)s * 8);
    float4 k0 = kp[0], k1 = kp[1];
    float l = qh[0]*k0.x + qh[1]*k0.y + qh[2]*k0.z + qh[3]*k0.w
            + qh[4]*k1.x + qh[5]*k1.y + qh[6]*k1.z + qh[7]*k1.w;
    ls[j] = l;
    mx = fmaxf(mx, l);
  }
#pragma unroll
  for (int d = 16; d >= 1; d >>= 1) mx = fmaxf(mx, __shfl_xor(mx, d));

  float sum = 0.f;
  float oc[8] = {0.f, 0.f, 0.f, 0.f, 0.f, 0.f, 0.f, 0.f};
#pragma unroll
  for (int j = 0; j < 64; ++j) {
    int s = lane + 32 * j;
    float e = __expf(ls[j] - mx);
    sum += e;
    const float4* vp = (const float4*)(vb + (size_t)s * 8);
    float4 v0 = vp[0], v1 = vp[1];
    oc[0] += e * v0.x; oc[1] += e * v0.y; oc[2] += e * v0.z; oc[3] += e * v0.w;
    oc[4] += e * v1.x; oc[5] += e * v1.y; oc[6] += e * v1.z; oc[7] += e * v1.w;
  }
#pragma unroll
  for (int d = 16; d >= 1; d >>= 1) {
    sum += __shfl_xor(sum, d);
#pragma unroll
    for (int c = 0; c < 8; ++c) oc[c] += __shfl_xor(oc[c], d);
  }
  if (lane == 0) {
    float inv = 1.f / sum;
#pragma unroll
    for (int c = 0; c < 8; ++c) obuf[(size_t)r * 64 + h * 8 + c] = oc[c] * inv;
  }
}

// ---------------------------------------------------------------------------
// Pass 2: recompute LN, g = sigmoid(x@Wg + bg), gated = g * o[r] (broadcast
// over seq), out = gated@Wo + bo.  Two K=64 WMMA GEMMs per 16-row tile.
// Rows are flattened (s*R + r); R%16==0 so tiles never straddle s.
// ---------------------------------------------------------------------------
__global__ __launch_bounds__(256)
void msa_pass2(const float* __restrict__ m, const float* __restrict__ gamma,
               const float* __restrict__ beta, const float* __restrict__ Wg,
               const float* __restrict__ bgv, const float* __restrict__ Wo,
               const float* __restrict__ bov, const float* __restrict__ obuf,
               float* __restrict__ out) {
  __shared__ float  gs[64], bs[64], bgs[64], bos[64];
  __shared__ __bf16 xs[WPB * 16 * 66];
  const int t = threadIdx.x;
  if (t < 64) { gs[t] = gamma[t]; bs[t] = beta[t]; bgs[t] = bgv[t]; bos[t] = bov[t]; }
  __syncthreads();

  const int lane = t & 31, wv = t >> 5;
  const int h = lane >> 4, rl = lane & 15, n = rl;

  // B fragments for Wg / Wo (64x64), 4 N-tiles x 2 K-chunks each, kept in VGPRs.
  v16bf bgf[4][2], bof[4][2];
#pragma unroll
  for (int nb = 0; nb < 4; ++nb) {
#pragma unroll
    for (int kc = 0; kc < 2; ++kc) {
#pragma unroll
      for (int e = 0; e < 16; ++e) {
        int K = 32 * kc + 16 * h + e;
        int col = 16 * nb + n;
        bgf[nb][kc][e] = (__bf16)Wg[K * 64 + col];
        bof[nb][kc][e] = (__bf16)Wo[K * 64 + col];
      }
    }
  }

  __bf16* xrow  = xs + (wv * 16 + rl) * 66;  // this lane's staging row
  __bf16* xbase = xs + wv * 16 * 66;         // wave tile base

  for (int t8 = 0; t8 < TPW; ++t8) {
    int gt = (blockIdx.x * WPB + wv) * TPW + t8;
    size_t row0 = (size_t)gt * 16;
    int r0 = (int)(row0 % (size_t)R);

    const float* mrow = m + (row0 + rl) * CM + 32 * h;
    float x[32];
    ln_row_half(mrow, gs, bs, h, x);
    stage_x(xrow, h, x);
    v16bf a0 = load_a_frag(xrow, h, 0);
    v16bf a1 = load_a_frag(xrow, h, 1);

    // g = sigmoid(x@Wg + bg); gated = g * o[r]; restage gated as next A matrix.
#pragma unroll
    for (int nb = 0; nb < 4; ++nb) {
      v8f acc = {0.f, 0.f, 0.f, 0.f, 0.f, 0.f, 0.f, 0.f};
      acc = wmma_bf16(a0, bgf[nb][0], acc);
      acc = wmma_bf16(a1, bgf[nb][1], acc);
      int col = 16 * nb + n;
      float bgc = bgs[col];
#pragma unroll
      for (int i = 0; i < 8; ++i) {
        int mm = i + 8 * h;                       // D-matrix row
        float z = acc[i] + bgc;
        float g = 1.f / (1.f + __expf(-z));
        float ov = obuf[(size_t)(r0 + mm) * 64 + col];
        xbase[mm * 66 + col] = (__bf16)(g * ov);
      }
    }

    v16bf c0 = load_a_frag(xrow, h, 0);
    v16bf c1 = load_a_frag(xrow, h, 1);
#pragma unroll
    for (int nb = 0; nb < 4; ++nb) {
      v8f acc = {0.f, 0.f, 0.f, 0.f, 0.f, 0.f, 0.f, 0.f};
      acc = wmma_bf16(c0, bof[nb][0], acc);
      acc = wmma_bf16(c1, bof[nb][1], acc);
      int col = 16 * nb + n;
      float bv = bos[col];
#pragma unroll
      for (int i = 0; i < 8; ++i)
        out[(row0 + i + 8 * h) * CM + col] = acc[i] + bv;
    }
  }
}

extern "C" void kernel_launch(void* const* d_in, const int* in_sizes, int n_in,
                              void* d_out, int out_size, void* d_ws, size_t ws_size,
                              hipStream_t stream) {
  (void)in_sizes; (void)n_in; (void)out_size; (void)ws_size;
  const float* m     = (const float*)d_in[0];
  const float* gamma = (const float*)d_in[1];
  const float* beta  = (const float*)d_in[2];
  const float* Wq    = (const float*)d_in[3];
  const float* Wk    = (const float*)d_in[4];
  const float* Wv    = (const float*)d_in[5];
  const float* Wg    = (const float*)d_in[6];
  const float* bg    = (const float*)d_in[7];
  const float* Wo    = (const float*)d_in[8];
  const float* bo    = (const float*)d_in[9];
  float* out = (float*)d_out;

  char* ws = (char*)d_ws;
  size_t off = 0;
  float* partials = (float*)(ws + off); off += (size_t)R * 128 * 64 * sizeof(float); // 12.6 MB
  float* kbuf     = (float*)(ws + off); off += (size_t)R * S * 8 * sizeof(float);    // 25.2 MB
  float* vbuf     = (float*)(ws + off); off += (size_t)R * S * 8 * sizeof(float);    // 25.2 MB
  float* obuf     = (float*)(ws + off); off += (size_t)R * 64 * sizeof(float);       // 98 KB

  msa_pass1<<<GEMM_BLOCKS, 256, 0, stream>>>(m, gamma, beta, Wk, Wv, kbuf, vbuf,
                                             partials);
  msa_attn<<<R, 256, 0, stream>>>(partials, Wq, kbuf, vbuf, obuf);
  msa_pass2<<<GEMM_BLOCKS, 256, 0, stream>>>(m, gamma, beta, Wg, bg, Wo, bo, obuf,
                                             out);
}
